// LSTM_15642270892552
// MI455X (gfx1250) — matrix operand order repaired and checked by reading, not compiled, not run on hardware
//
#include <hip/hip_runtime.h>
#include <hip/hip_bf16.h>

#define B_DIM 256
#define T_DIM 512
#define I_DIM 256
#define H_DIM 256
#define K_DIM (I_DIM + H_DIM)   // 512
#define G4H   (4 * H_DIM)       // 1024

typedef __attribute__((ext_vector_type(16))) __bf16 v16bf;
typedef __attribute__((ext_vector_type(8)))  __bf16 v8bf;
typedef __attribute__((ext_vector_type(8)))  float  v8f;
typedef __attribute__((ext_vector_type(4)))  float  f4;

static __device__ inline unsigned short bf16_bits(float f) {
    __bf16 b = (__bf16)f;
    return __builtin_bit_cast(unsigned short, b);
}

static __device__ inline v16bf cat8(v8bf lo, v8bf hi) {
    return __builtin_shufflevector(lo, hi, 0,1,2,3,4,5,6,7,8,9,10,11,12,13,14,15);
}

// Convert 8 consecutive fp32 (two float4) to v8bf
static __device__ inline v8bf cvt8(f4 a, f4 b) {
    v8bf r;
    r[0]=(__bf16)a.x; r[1]=(__bf16)a.y; r[2]=(__bf16)a.z; r[3]=(__bf16)a.w;
    r[4]=(__bf16)b.x; r[5]=(__bf16)b.y; r[6]=(__bf16)b.z; r[7]=(__bf16)b.w;
    return r;
}

static __device__ inline float fast_sigmoid(float x) {
    return 1.0f / (1.0f + __expf(-x));
}

// ---------------- one-time prep kernels ----------------

__global__ __launch_bounds__(256)
void cvt_w_kernel(const float* __restrict__ W, unsigned short* __restrict__ Wbf, int n) {
    int i = blockIdx.x * 256 + threadIdx.x;
    if (i < n) Wbf[i] = bf16_bits(W[i]);
}

__global__ __launch_bounds__(256)
void init_state_kernel(unsigned short* __restrict__ hbf, float* __restrict__ c, int n) {
    int i = blockIdx.x * 256 + threadIdx.x;
    if (i < n) { hbf[i] = 0; c[i] = 0.0f; }
}

// ---------------- per-timestep fused GEMM + LSTM cell ----------------
// Each wave: 16(batch) x 16(hidden) tile, all 4 gates.
// All 8 waves of a block share one hidden-column base nb, so the block's
// 64 KB W tile (4 gates x 16 rows x 512 K, bf16) is staged into LDS once
// via GLOBAL_LOAD_ASYNC_TO_LDS_B128 and consumed by all waves via ds loads.
// Grid: 256 wave-tiles = 32 blocks x 8 waves.

__global__ __launch_bounds__(256)
void lstm_step_kernel(const float* __restrict__ x,           // [B, T, I] fp32
                      const unsigned short* __restrict__ Wbf,// [4H, K] bf16 bits, row-major
                      const float* __restrict__ bias,        // [4H]
                      unsigned short* __restrict__ hbf,      // [B, H] bf16 bits (state)
                      float* __restrict__ c,                 // [B, H] fp32 (state)
                      float* __restrict__ hout,              // [B, H] fp32 (d_out)
                      int t, int write_out)
{
    extern __shared__ unsigned short smem[];           // 64 KB: [64 rows][512 halves]
                                                       // row = g*16 + r  (r = hidden col in tile)

    const int tid  = threadIdx.x;
    const int lane = tid & 31;
    const int wave = tid >> 5;                         // 0..7
    const int tile = blockIdx.x * 8 + wave;            // 0..255
    const int mb   = (tile & 15) << 4;                 // batch tile base (per wave)
    const int nb   = (int)(blockIdx.x >> 1) << 4;      // hidden tile base (same for whole block)
    const int nl   = lane & 15;                        // N (and A-row M) index
    const int half = lane >> 4;                        // lane half

    // ---- stage the block's W tile into LDS (async, b128 per lane) ----
    // 64 KB = 4096 16-byte chunks; 256 threads x 16 iterations.
    #pragma unroll
    for (int it = 0; it < 16; ++it) {
        const int chunk = it * 256 + tid;              // 0..4095
        const int lrow  = chunk >> 6;                  // 0..63  (g*16 + r)
        const int cc    = (chunk & 63) * 8;            // half-element column start
        const int g     = lrow >> 4;
        const int r     = lrow & 15;
        const unsigned short* gsrc =
            Wbf + ((size_t)(g * H_DIM + nb + r)) * K_DIM + cc;
        const unsigned lds_off = (unsigned)(chunk * 16);   // dynamic LDS starts at offset 0
        asm volatile("global_load_async_to_lds_b128 %0, %1, off"
                     :: "v"(lds_off), "v"(gsrc) : "memory");
    }
    asm volatile("s_wait_asynccnt 0x0" ::: "memory");
    __syncthreads();

    v8f acc0 = {}, acc1 = {}, acc2 = {}, acc3 = {};    // gates i, j, f, o
    const int arow = mb + nl;                          // this lane's A-matrix row (batch)

    for (int k0 = 0; k0 < K_DIM; k0 += 32) {
        // ---- A fragment: catted[arow, k0 .. k0+31] (bf16) ----
        v16bf afrag;
        if (k0 < I_DIM) {
            // from fp32 x, convert in-register
            const float* px = x + ((size_t)arow * T_DIM + t) * I_DIM + k0 + half * 8;
            const f4* p0 = (const f4*)px;          // K = k0+half*8 + 0..7
            const f4* p1 = (const f4*)(px + 16);   // K = k0+16+half*8 + 0..7
            v8bf lo = cvt8(p0[0], p0[1]);
            v8bf hi = cvt8(p1[0], p1[1]);
            afrag = cat8(lo, hi);
        } else {
            // from bf16 h state
            const unsigned short* ph = hbf + (size_t)arow * H_DIM + (k0 - I_DIM) + half * 8;
            v8bf lo = *(const v8bf*)ph;            // K offsets 0..7
            v8bf hi = *(const v8bf*)(ph + 16);     // K offsets 16..23
            afrag = cat8(lo, hi);
        }

        // ---- B fragments from LDS: smem[g*16 + nl][k0 + half*8 + 0..7, k0 + 16 + half*8 + 0..7] ----
        const int kcol = k0 + half * 8;
        const unsigned short* pw0 = smem + ((size_t)( 0 + nl)) * K_DIM + kcol;
        const unsigned short* pw1 = smem + ((size_t)(16 + nl)) * K_DIM + kcol;
        const unsigned short* pw2 = smem + ((size_t)(32 + nl)) * K_DIM + kcol;
        const unsigned short* pw3 = smem + ((size_t)(48 + nl)) * K_DIM + kcol;
        v16bf b0 = cat8(((const v8bf*)pw0)[0], ((const v8bf*)pw0)[2]);
        v16bf b1 = cat8(((const v8bf*)pw1)[0], ((const v8bf*)pw1)[2]);
        v16bf b2 = cat8(((const v8bf*)pw2)[0], ((const v8bf*)pw2)[2]);
        v16bf b3 = cat8(((const v8bf*)pw3)[0], ((const v8bf*)pw3)[2]);

        acc0 = __builtin_amdgcn_wmma_f32_16x16x32_bf16(false, afrag, false, b0,
                                                       (short)0, acc0, false, false);
        acc1 = __builtin_amdgcn_wmma_f32_16x16x32_bf16(false, afrag, false, b1,
                                                       (short)0, acc1, false, false);
        acc2 = __builtin_amdgcn_wmma_f32_16x16x32_bf16(false, afrag, false, b2,
                                                       (short)0, acc2, false, false);
        acc3 = __builtin_amdgcn_wmma_f32_16x16x32_bf16(false, afrag, false, b3,
                                                       (short)0, acc3, false, false);
    }

    // ---- epilogue: gate activations + recurrence ----
    const int n  = nb + nl;
    const float bi = bias[0 * H_DIM + n];
    const float bj = bias[1 * H_DIM + n];
    const float bf = bias[2 * H_DIM + n];
    const float bo = bias[3 * H_DIM + n];

    #pragma unroll
    for (int v = 0; v < 8; ++v) {
        const int row = mb + v + 8 * half;             // D layout: M = v + 8*half
        const size_t idx = (size_t)row * H_DIM + n;
        const float ai = fast_sigmoid(acc0[v] + bi);
        const float aj = tanhf       (acc1[v] + bj);
        const float af = fast_sigmoid(acc2[v] + bf);
        const float ao = fast_sigmoid(acc3[v] + bo);
        const float cn = ai * aj + c[idx] * af;
        const float hn = ao * tanhf(cn);
        c[idx]   = cn;
        hbf[idx] = bf16_bits(hn);
        if (write_out) hout[idx] = hn;
    }
}

// ---------------- host launcher ----------------

extern "C" void kernel_launch(void* const* d_in, const int* in_sizes, int n_in,
                              void* d_out, int out_size, void* d_ws, size_t ws_size,
                              hipStream_t stream) {
    const float* x  = (const float*)d_in[0];   // [B, T, I]
    const float* W  = (const float*)d_in[1];   // [4H, I+H]
    const float* bb = (const float*)d_in[2];   // [4H]
    float* out = (float*)d_out;                // [B, H]

    // workspace layout: Wbf [4H*K] bf16 | hbf [B*H] bf16 | c [B*H] fp32  (~1.4 MB)
    unsigned short* Wbf = (unsigned short*)d_ws;
    unsigned short* hbf = Wbf + (size_t)G4H * K_DIM;
    float* c = (float*)(hbf + (size_t)B_DIM * H_DIM);

    const int wn = G4H * K_DIM;                 // 524288
    cvt_w_kernel<<<(wn + 255) / 256, 256, 0, stream>>>(W, Wbf, wn);

    const int sn = B_DIM * H_DIM;               // 65536
    init_state_kernel<<<(sn + 255) / 256, 256, 0, stream>>>(hbf, c, sn);

    // 256 wave-tiles (16x16 over BxH), 8 waves/block -> 32 blocks of 256 threads,
    // 64 KB dynamic LDS per block for the staged W tile.
    for (int t = 0; t < T_DIM; ++t) {
        lstm_step_kernel<<<32, 256, 65536, stream>>>(x, Wbf, bb, hbf, c, out,
                                                     t, (t == T_DIM - 1) ? 1 : 0);
    }
}
